// GT_58798102282916
// MI455X (gfx1250) — matrix operand order repaired and checked by reading, not compiled, not run on hardware
//
#include <hip/hip_runtime.h>
#include <hip/hip_bf16.h>
#include <math.h>

// ---------------------------------------------------------------------------
// Graph-transformer VGAE forward for MI455X (gfx1250, wave32, WMMA).
//  - GEMMs: v_wmma_f32_16x16x32_bf16, 5 row-tiles per wave (80 rows/block),
//    double-buffered LDS A-tiles filled with GLOBAL_LOAD_ASYNC_TO_LDS_B128
//    (ASYNCcnt), one barrier per K-step.
//  - Activations kept in bf16 in memory (q/k/v gathers are bf16 -> half the
//    L2 gather traffic; 50000x128 bf16 = 12.8MB per matrix, L2-resident).
//  - Edge softmax: u32-encoded float atomicMax + hardware f32 atomicAdd.
// ---------------------------------------------------------------------------

#define GT_N 50000
#define GT_E 800000
#define GT_HEADS 4
#define GEMM_ROWS 80   // 50000 = 625 * 80; 5 row tiles of 16 per block

typedef __bf16 v16bf __attribute__((ext_vector_type(16)));
typedef float  v8f   __attribute__((ext_vector_type(8)));

union Frag {
    v16bf v;
    uint4 q[2];
    unsigned int u[8];
};

// ---- helpers --------------------------------------------------------------

__device__ __forceinline__ unsigned short f2bf(float f) {
    unsigned int u = __float_as_uint(f);
    unsigned int r = (u + 0x7FFFu + ((u >> 16) & 1u)) >> 16;  // RNE
    return (unsigned short)r;
}
__device__ __forceinline__ float bf2f(unsigned int h) {
    return __uint_as_float(h << 16);
}

// monotone float <-> uint encoding so unsigned atomicMax == float max
__device__ __forceinline__ unsigned int enc_f32(float f) {
    unsigned int u = __float_as_uint(f);
    return u ^ ((unsigned int)((int)u >> 31) | 0x80000000u);
}
__device__ __forceinline__ float dec_f32(unsigned int u) {
    unsigned int v = (u & 0x80000000u) ? (u ^ 0x80000000u) : ~u;
    return __uint_as_float(v);
}

__device__ __forceinline__ unsigned int hash_u32(unsigned int x) {
    x ^= x >> 16; x *= 0x7feb352dU;
    x ^= x >> 15; x *= 0x846ca68bU;
    x ^= x >> 16;
    return x;
}

__device__ __forceinline__ void wait_async0() {
#if __has_builtin(__builtin_amdgcn_s_wait_asynccnt)
    __builtin_amdgcn_s_wait_asynccnt(0);
#else
    asm volatile("s_wait_asynccnt 0x0" ::: "memory");
#endif
}

__device__ __forceinline__ void async_copy16(unsigned lds_off,
                                             const unsigned short* gptr) {
    unsigned long long g = (unsigned long long)gptr;
    asm volatile("global_load_async_to_lds_b128 %0, %1, off"
                 :: "v"(lds_off), "v"(g) : "memory");
}

// ---- f32 -> bf16 conversion ----------------------------------------------

__global__ void k_f32_to_bf16(const float* __restrict__ in,
                              unsigned short* __restrict__ out, int n) {
    int i = blockIdx.x * blockDim.x + threadIdx.x;
    if (i < n) out[i] = f2bf(in[i]);
}

// ---- WMMA GEMM: C[N x M] = A[N x K](bf16) * W[K x M](bf16) ----------------
// grid.x = N/80, grid.z selects weight/output slice (Q,K,V).
// blockDim.x = 32 * (M/16): one wave per 16-wide column tile, 5 row tiles.

__global__ void k_wmma_gemm(const unsigned short* __restrict__ Abf,
                            const unsigned short* __restrict__ Wbf,
                            float* __restrict__ Cf,          // optional
                            unsigned short* __restrict__ Cbf, // optional
                            int K, int M, int do_relu,
                            long wstride, long cstride) {
    const int z = blockIdx.z;
    const unsigned short* W = Wbf + (long)z * wstride;

    const int tid   = threadIdx.x;
    const int lane  = tid & 31;
    const int wave  = tid >> 5;
    const int half  = lane >> 4;     // 0: lanes 0-15, 1: lanes 16-31
    const int m_in  = lane & 15;
    const long rowB = (long)blockIdx.x * GEMM_ROWS;
    const int nBase = wave * 16;

    __shared__ unsigned short tile[2][GEMM_ROWS * 32];  // 2 x 5KB, bf16

    v8f zero8 = {0.f, 0.f, 0.f, 0.f, 0.f, 0.f, 0.f, 0.f};
    v8f acc[5];
#pragma unroll
    for (int rt = 0; rt < 5; ++rt) acc[rt] = zero8;

    // issue async copies of one 80x32 bf16 tile (16B chunks, 4 per row)
    auto issue_tile = [&](int kb, int buf) {
        for (int c = tid; c < GEMM_ROWS * 4; c += blockDim.x) {
            int row = c >> 2;
            int seg = c & 3;
            const unsigned short* g = Abf + (rowB + row) * K + kb + seg * 8;
            unsigned loff = (unsigned)(size_t)&tile[buf][row * 32 + seg * 8];
            async_copy16(loff, g);
        }
    };

    issue_tile(0, 0);
    wait_async0();
    __syncthreads();

    int cur = 0;
    for (int kb = 0; kb < K; kb += 32, cur ^= 1) {
        if (kb + 32 < K) issue_tile(kb + 32, cur ^ 1);

        // B fragment: lane holds column n = nBase+m_in, 16 k values in table order
        Frag b;
        {
            const unsigned short* wp = W + (long)kb * M + nBase + m_in;
#pragma unroll
            for (int j = 0; j < 8; ++j) {
                int k0 = 8 * half + 2 * j + ((j >= 4) ? 8 : 0);
                b.u[j] = (unsigned int)wp[(long)k0 * M] | ((unsigned int)wp[(long)(k0 + 1) * M] << 16);
            }
        }

        const unsigned int* tu = (const unsigned int*)tile[cur];
#pragma unroll
        for (int rt = 0; rt < 5; ++rt) {
            // A fragment per ISA 16-bit A 16x32 layout:
            // lanes 0-15: k 0..7 & 16..23 ; lanes 16-31: k 8..15 & 24..31
            Frag a;
            const uint4* lp = (const uint4*)(tu + (rt * 16 + m_in) * 16);
            a.q[0] = lp[half];
            a.q[1] = lp[2 + half];
            acc[rt] = __builtin_amdgcn_wmma_f32_16x16x32_bf16(
                false, a.v, false, b.v, (short)0, acc[rt], false, false);
        }

        if (kb + 32 < K) wait_async0();
        __syncthreads();
    }

    // C/D layout: lane l, vgpr r -> (row = rt*16 + r + half*8, col = lane&15)
#pragma unroll
    for (int rt = 0; rt < 5; ++rt) {
#pragma unroll
        for (int r = 0; r < 8; ++r) {
            float v = acc[rt][r];
            if (do_relu) v = fmaxf(v, 0.f);
            long row = rowB + rt * 16 + r + half * 8;
            if (Cf)  (Cf + (long)z * cstride)[row * M + nBase + m_in] = v;
            if (Cbf) (Cbf + (long)z * cstride)[row * M + nBase + m_in] = f2bf(v);
        }
    }
}

// ---- edge attention (q,k,v are bf16) --------------------------------------

__global__ void k_edge_scores(const unsigned short* __restrict__ q,
                              const unsigned short* __restrict__ k,
                              const int* __restrict__ src,
                              const int* __restrict__ dst,
                              float* __restrict__ sbuf,
                              unsigned int* __restrict__ menc,
                              int nE, int M, int dh, float scale) {
    int t = blockIdx.x * blockDim.x + threadIdx.x;
    if (t >= nE * GT_HEADS) return;
    int e  = t >> 2;
    int hh = t & 3;
    int d  = dst[e];
    int s  = src[e];
    const uint4* qp = (const uint4*)(q + (long)d * M + hh * dh);
    const uint4* kp = (const uint4*)(k + (long)s * M + hh * dh);
    float acc = 0.f;
    for (int i = 0; i < (dh >> 3); ++i) {
        uint4 a = qp[i], b = kp[i];
        acc += bf2f(a.x & 0xffffu) * bf2f(b.x & 0xffffu) + bf2f(a.x >> 16) * bf2f(b.x >> 16);
        acc += bf2f(a.y & 0xffffu) * bf2f(b.y & 0xffffu) + bf2f(a.y >> 16) * bf2f(b.y >> 16);
        acc += bf2f(a.z & 0xffffu) * bf2f(b.z & 0xffffu) + bf2f(a.z >> 16) * bf2f(b.z >> 16);
        acc += bf2f(a.w & 0xffffu) * bf2f(b.w & 0xffffu) + bf2f(a.w >> 16) * bf2f(b.w >> 16);
    }
    acc *= scale;
    sbuf[t] = acc;
    atomicMax(menc + d * GT_HEADS + hh, enc_f32(acc));
}

__global__ void k_edge_accum(const unsigned short* __restrict__ v,
                             const int* __restrict__ src,
                             const int* __restrict__ dst,
                             const float* __restrict__ sbuf,
                             const unsigned int* __restrict__ menc,
                             float* __restrict__ den,
                             float* __restrict__ agg,
                             int nE, int M, int dh) {
    int t = blockIdx.x * blockDim.x + threadIdx.x;
    if (t >= nE * GT_HEADS) return;
    int e  = t >> 2;
    int hh = t & 3;
    int d  = dst[e];
    int s  = src[e];
    float m  = dec_f32(menc[d * GT_HEADS + hh]);
    float ex = __expf(sbuf[t] - m);
    atomicAdd(den + d * GT_HEADS + hh, ex);
    const uint4* vp = (const uint4*)(v + (long)s * M + hh * dh);
    float* ap = agg + (long)d * M + hh * dh;
    for (int i = 0; i < (dh >> 3); ++i) {
        uint4 vv = vp[i];
        atomicAdd(ap + i * 8 + 0, ex * bf2f(vv.x & 0xffffu));
        atomicAdd(ap + i * 8 + 1, ex * bf2f(vv.x >> 16));
        atomicAdd(ap + i * 8 + 2, ex * bf2f(vv.y & 0xffffu));
        atomicAdd(ap + i * 8 + 3, ex * bf2f(vv.y >> 16));
        atomicAdd(ap + i * 8 + 4, ex * bf2f(vv.z & 0xffffu));
        atomicAdd(ap + i * 8 + 5, ex * bf2f(vv.z >> 16));
        atomicAdd(ap + i * 8 + 6, ex * bf2f(vv.w & 0xffffu));
        atomicAdd(ap + i * 8 + 7, ex * bf2f(vv.w >> 16));
    }
}

// alpha normalization hoisted to per-node pass; emits bf16 for next GEMM
__global__ void k_finalize(const float* __restrict__ agg,
                           const float* __restrict__ den,
                           unsigned short* __restrict__ aggbf,
                           int n, int M, int dh) {
    int idx = blockIdx.x * blockDim.x + threadIdx.x;
    if (idx >= n * M) return;
    int node = idx / M;
    int col  = idx - node * M;
    int hh   = col / dh;
    float dn = den[node * GT_HEADS + hh];
    float v  = (dn > 0.f) ? agg[idx] / dn : 0.f;
    aggbf[idx] = f2bf(v);
}

// ---- post: L2 norm (+ optional relu+dropout); f32 in, f32/bf16 out --------
// one wave per node, D in {64,128}

__global__ void k_l2norm_post(const float* __restrict__ in,
                              float* __restrict__ outf,          // optional
                              unsigned short* __restrict__ outbf, // optional
                              int n, int D, int mode, unsigned int seed) {
    int wid  = (blockIdx.x * blockDim.x + threadIdx.x) >> 5;
    int lane = threadIdx.x & 31;
    if (wid >= n) return;
    float vals[4];
    int nv = D >> 5;
    float ss = 0.f;
    for (int j = 0; j < nv; ++j) {
        float x = in[(long)wid * D + lane + 32 * j];
        vals[j] = x;
        ss += x * x;
    }
    for (int off = 16; off; off >>= 1) ss += __shfl_xor(ss, off, 32);
    float inv = 1.f / fmaxf(sqrtf(ss), 1e-12f);
    for (int j = 0; j < nv; ++j) {
        float y = vals[j] * inv;
        if (mode) {  // relu + dropout(p=0.5), scale by 2
            y = fmaxf(y, 0.f);
            unsigned int h = hash_u32(seed ^ (unsigned int)(wid * D + lane + 32 * j) * 2654435761u);
            y = (h & 0x80000000u) ? y * 2.f : 0.f;
        }
        long o = (long)wid * D + lane + 32 * j;
        if (outf)  outf[o]  = y;
        if (outbf) outbf[o] = f2bf(y);
    }
}

// ---- reparameterize: z = eps * (exp(logvar)+1e-12) + mu -------------------

__global__ void k_reparam(const float* __restrict__ mu,
                          const float* __restrict__ lv,
                          float* __restrict__ out, int n) {
    int idx = blockIdx.x * blockDim.x + threadIdx.x;
    if (idx >= n) return;
    float m  = mu[idx];
    float sd = __expf(lv[idx]) + 1e-12f;
    unsigned int h1 = hash_u32((unsigned int)idx * 2u + 0x9e3779b9u);
    unsigned int h2 = hash_u32((unsigned int)idx * 2u + 0x85ebca6bu);
    float u1 = ((float)((h1 >> 8) + 1u)) * (1.f / 16777216.f);   // (0,1]
    float u2 = ((float)(h2 >> 8)) * (1.f / 16777216.f);
    float r  = sqrtf(-2.f * __logf(u1));
    float eps = r * __cosf(6.28318530718f * u2);
    out[idx] = eps * sd + m;
}

// ---------------------------------------------------------------------------
// Host-side orchestration
// ---------------------------------------------------------------------------

static inline int cdiv(int a, int b) { return (a + b - 1) / b; }

static void run_gt_block(const unsigned short* Xin_bf, int K0, int M,
                         const float* wqkv0, const float* wqkvr, const float* wo,
                         float* h_f32, unsigned short* h_bf,
                         unsigned short* qkv_bf, float* agg, unsigned short* agg_bf,
                         float* sbuf, unsigned int* menc, float* den,
                         unsigned short* wbf_qkv, unsigned short* wbf_o,
                         const int* src, const int* dst, hipStream_t stream) {
    const int N = GT_N, E = GT_E;
    const int dh = M / GT_HEADS;
    const float scale = 1.f / sqrtf((float)dh);
    const int tE = E * GT_HEADS;
    const int bdim = 32 * (M / 16);

    for (int l = 0; l < 3; ++l) {
        int K = (l == 0) ? K0 : M;
        const float* w = (l == 0) ? wqkv0 : (wqkvr + (size_t)(l - 1) * 3 * M * M);
        int wcnt = 3 * K * M;

        k_f32_to_bf16<<<cdiv(wcnt, 256), 256, 0, stream>>>(w, wbf_qkv, wcnt);
        k_f32_to_bf16<<<cdiv(M * M, 256), 256, 0, stream>>>(wo + (size_t)l * M * M, wbf_o, M * M);

        const unsigned short* A = (l == 0) ? Xin_bf : h_bf;
        // Q,K,V in one launch via grid.z; bf16 outputs only
        k_wmma_gemm<<<dim3(N / GEMM_ROWS, 1, 3), bdim, 0, stream>>>(
            A, wbf_qkv, (float*)nullptr, qkv_bf, K, M, 0, (long)K * M, (long)N * M);

        hipMemsetAsync(menc, 0, (size_t)N * GT_HEADS * sizeof(unsigned int), stream);
        hipMemsetAsync(den, 0, (size_t)N * GT_HEADS * sizeof(float), stream);
        hipMemsetAsync(agg, 0, (size_t)N * M * sizeof(float), stream);

        k_edge_scores<<<cdiv(tE, 256), 256, 0, stream>>>(
            qkv_bf, qkv_bf + (size_t)N * M, src, dst, sbuf, menc, E, M, dh, scale);
        k_edge_accum<<<cdiv(tE, 256), 256, 0, stream>>>(
            qkv_bf + 2ull * N * M, src, dst, sbuf, menc, den, agg, E, M, dh);
        k_finalize<<<cdiv(N * M, 256), 256, 0, stream>>>(agg, den, agg_bf, N, M, dh);

        // H = agg @ wo  (f32 + bf16 outputs; relu except last layer)
        k_wmma_gemm<<<dim3(N / GEMM_ROWS, 1, 1), bdim, 0, stream>>>(
            agg_bf, wbf_o, h_f32, h_bf, M, M, (l < 2) ? 1 : 0, 0, 0);
    }
}

extern "C" void kernel_launch(void* const* d_in, const int* in_sizes, int n_in,
                              void* d_out, int out_size, void* d_ws, size_t ws_size,
                              hipStream_t stream) {
    const int N = GT_N;
    const int E = GT_E;

    const float* feat    = (const float*)d_in[0];
    const int*   src     = (const int*)d_in[1];
    const int*   dst     = (const int*)d_in[2];
    const float* wqkv1_0 = (const float*)d_in[3];
    const float* wqkv1_r = (const float*)d_in[4];
    const float* wo1     = (const float*)d_in[5];
    const float* wqkv2_0 = (const float*)d_in[6];
    const float* wqkv2_r = (const float*)d_in[7];
    const float* wo2     = (const float*)d_in[8];
    const float* wqkv3_0 = (const float*)d_in[9];
    const float* wqkv3_r = (const float*)d_in[10];
    const float* wo3     = (const float*)d_in[11];
    float* out = (float*)d_out;

    // workspace carve-out (~194 MB), all regions 16B-aligned
    float* wsf = (float*)d_ws;
    float* h_f32 = wsf;                                        // N*128
    float* agg   = h_f32 + (size_t)N * 128;                    // N*128
    float* sbuf  = agg + (size_t)N * 128;                      // E*4
    float* den   = sbuf + (size_t)E * 4;                       // N*4
    unsigned int* menc = (unsigned int*)(den + (size_t)N * 4); // N*4
    float* mubuf = (float*)(menc + (size_t)N * 4);             // N*64
    float* lvbuf = mubuf + (size_t)N * 64;                     // N*64
    unsigned short* feat_bf = (unsigned short*)(lvbuf + (size_t)N * 64); // N*256
    unsigned short* h_bf    = feat_bf + (size_t)N * 256;       // N*128
    unsigned short* x_bf    = h_bf + (size_t)N * 128;          // N*128
    unsigned short* qkv_bf  = x_bf + (size_t)N * 128;          // 3*N*128
    unsigned short* agg_bf  = qkv_bf + 3ull * N * 128;         // N*128
    unsigned short* wbf_qkv = agg_bf + (size_t)N * 128;        // 3*256*128
    unsigned short* wbf_o   = wbf_qkv + 3 * 256 * 128;         // 128*128

    // features -> bf16 once
    k_f32_to_bf16<<<cdiv(N * 256, 256), 256, 0, stream>>>(feat, feat_bf, N * 256);

    // gc1: 256 -> 128 (3 layers)
    run_gt_block(feat_bf, 256, 128, wqkv1_0, wqkv1_r, wo1,
                 h_f32, h_bf, qkv_bf, agg, agg_bf, sbuf, menc, den,
                 wbf_qkv, wbf_o, src, dst, stream);
    // x = dropout(relu(l2norm(h)))  -> bf16 feed for gc2/gc3
    k_l2norm_post<<<cdiv(N * 32, 256), 256, 0, stream>>>(
        h_f32, (float*)nullptr, x_bf, N, 128, 1, 0x1234567u);

    // gc2: 128 -> 64 => mu = l2norm(...)
    run_gt_block(x_bf, 128, 64, wqkv2_0, wqkv2_r, wo2,
                 h_f32, h_bf, qkv_bf, agg, agg_bf, sbuf, menc, den,
                 wbf_qkv, wbf_o, src, dst, stream);
    k_l2norm_post<<<cdiv(N * 32, 256), 256, 0, stream>>>(
        h_f32, mubuf, (unsigned short*)nullptr, N, 64, 0, 0);

    // gc3: 128 -> 64 => logvar = l2norm(...)
    run_gt_block(x_bf, 128, 64, wqkv3_0, wqkv3_r, wo3,
                 h_f32, h_bf, qkv_bf, agg, agg_bf, sbuf, menc, den,
                 wbf_qkv, wbf_o, src, dst, stream);
    k_l2norm_post<<<cdiv(N * 32, 256), 256, 0, stream>>>(
        h_f32, lvbuf, (unsigned short*)nullptr, N, 64, 0, 0);

    // z = eps * (exp(logvar)+1e-12) + mu
    k_reparam<<<cdiv(N * 64, 256), 256, 0, stream>>>(mubuf, lvbuf, out, N * 64);
}
